// SemanticAlignmentModule_30520037605542
// MI455X (gfx1250) — matrix operand order repaired and checked by reading, not compiled, not run on hardware
//
#include <hip/hip_runtime.h>
#include <hip/hip_bf16.h>

typedef _Float16 v16h __attribute__((ext_vector_type(16)));
typedef _Float16 v8h  __attribute__((ext_vector_type(8)));
typedef float    v8f  __attribute__((ext_vector_type(8)));

// ---------------- problem constants ----------------
#define BATCH     8
#define HW        4096            // 64*64
#define ROWS      32768           // BATCH*HW
#define CV        512
#define TT        77
#define HEADS     8
#define DH        64
#define FF        2048
#define KV_ROWS   (BATCH*TT)      // 616

__device__ __forceinline__ float wave_sum(float v) {
    for (int m = 16; m > 0; m >>= 1) v += __shfl_xor(v, m, 32);
    return v;
}

// ---------------- weight transpose + f16 convert: dst[n*K+k] = src[k*N+n] ----
__global__ void transpose_cvt_kernel(const float* __restrict__ src,
                                     _Float16* __restrict__ dst, int K, int Nn) {
    size_t idx = (size_t)blockIdx.x * blockDim.x + threadIdx.x;
    size_t total = (size_t)K * Nn;
    if (idx >= total) return;
    int n = (int)(idx / K);
    int k = (int)(idx % K);
    dst[idx] = (_Float16)src[(size_t)k * Nn + n];
}

__global__ void cvt_kernel(const float* __restrict__ src,
                           _Float16* __restrict__ dst, int count) {
    int idx = blockIdx.x * blockDim.x + threadIdx.x;
    if (idx < count) dst[idx] = (_Float16)src[idx];
}

// ---------------- pad mask over text rows -----------------------------------
__global__ __launch_bounds__(128)
void padmask_kernel(const float* __restrict__ text, int* __restrict__ mask) {
    int row = blockIdx.x;                     // 0..615
    float s = 0.f;
    for (int i = threadIdx.x; i < CV; i += 128)
        s += fabsf(text[(size_t)row * CV + i]);
    s = wave_sum(s);
    __shared__ float ps[4];
    if ((threadIdx.x & 31) == 0) ps[threadIdx.x >> 5] = s;
    __syncthreads();
    if (threadIdx.x == 0)
        mask[row] = ((ps[0] + ps[1] + ps[2] + ps[3]) <= 1e-6f) ? 1 : 0;
}

__global__ void allpad_kernel(const int* __restrict__ mask, int* __restrict__ allpad) {
    int b = threadIdx.x;
    if (b >= BATCH) return;
    int all = 1;
    for (int t = 0; t < TT; t++) all &= mask[b * TT + t];
    allpad[b] = all;
}

// ---------------- LayerNorm: row of 512, writes f32 + f16 -------------------
__global__ __launch_bounds__(128)
void layernorm_kernel(const float* __restrict__ in, const float* __restrict__ g,
                      const float* __restrict__ bvec, float* __restrict__ outF,
                      _Float16* __restrict__ outH) {
    const int row = blockIdx.x;
    const int tid = threadIdx.x;
    const float4 x4 = ((const float4*)(in + (size_t)row * CV))[tid];
    float s  = x4.x + x4.y + x4.z + x4.w;
    float sq = x4.x * x4.x + x4.y * x4.y + x4.z * x4.z + x4.w * x4.w;
    s = wave_sum(s); sq = wave_sum(sq);
    __shared__ float ps[4], pq[4];
    if ((tid & 31) == 0) { ps[tid >> 5] = s; pq[tid >> 5] = sq; }
    __syncthreads();
    s  = ps[0] + ps[1] + ps[2] + ps[3];
    sq = pq[0] + pq[1] + pq[2] + pq[3];
    const float mean = s * (1.0f / CV);
    const float var  = sq * (1.0f / CV) - mean * mean;
    const float rstd = rsqrtf(var + 1e-5f);
    const int c0 = tid * 4;
    float o[4] = {x4.x, x4.y, x4.z, x4.w};
    for (int i = 0; i < 4; i++) {
        float val = (o[i] - mean) * rstd * g[c0 + i] + bvec[c0 + i];
        outF[(size_t)row * CV + c0 + i] = val;
        outH[(size_t)row * CV + c0 + i] = (_Float16)val;
    }
}

// ---------------- per-head L2 norm of K rows (in place) ---------------------
__global__ __launch_bounds__(256)
void l2norm_heads_kernel(float* __restrict__ x, int units) {
    int unit = blockIdx.x * 8 + (threadIdx.x >> 5);
    int lane = threadIdx.x & 31;
    if (unit >= units) return;
    size_t base = (size_t)unit * DH;
    float a = x[base + lane * 2];
    float c = x[base + lane * 2 + 1];
    float ss = wave_sum(a * a + c * c);
    float iv = 1.0f / fmaxf(sqrtf(ss), 1e-6f);
    x[base + lane * 2]     = a * iv;
    x[base + lane * 2 + 1] = c * iv;
}

// ---------------- WMMA GEMM: C = A(MxK,f16) * Bt(NxK,f16)^T + epilogue ------
// Double-buffered LDS (ping-pong), one barrier per K-step, global prefetch of
// tile i+2 while tile i+1 loads land under the WMMA block.
// EPI 0: outF = acc + bias
// EPI 1: outF = resid + alpha*(acc + bias)
// EPI 2: outH = gelu(acc + bias)        (exact erf gelu)
// EPI 3: outF = resid + acc + bias
template <int EPI>
__global__ __launch_bounds__(256)
void gemm_wmma_f16(const _Float16* __restrict__ A, const _Float16* __restrict__ Bt,
                   const float* __restrict__ bias, const float* __restrict__ resid,
                   const float* __restrict__ alphap, float* __restrict__ outF,
                   _Float16* __restrict__ outH, int M, int Nn, int K) {
    constexpr int BM = 128, BN = 128, BK = 32, LDB = BK + 8;
    __shared__ __align__(16) _Float16 As[2][BM * LDB];
    __shared__ __align__(16) _Float16 Bs[2][BN * LDB];

    const int tid   = threadIdx.x;
    const int lane  = tid & 31;
    const int wave  = tid >> 5;
    const int wmBase = (wave >> 2) * 64;   // 2 waves along M
    const int wnBase = (wave & 3) * 32;    // 4 waves along N
    const int blockN = blockIdx.x * BN;
    const int blockM = blockIdx.y * BM;

    const int lr = lane & 15;
    const int hs = lane >> 4;              // half-select
    const int kA0 = hs * 8;                // A chunk0: K {0..7}/{8..15}
    const int kA1 = 16 + hs * 8;           // A chunk1: K {16..23}/{24..31}
    const int kB0 = hs * 8;                // B: K {0..7,16..23}/{8..15,24..31} (matches A)

    const int ldRow = tid >> 1;            // 0..127
    const int ldCol = (tid & 1) * 16;      // 0 or 16
    const int aRow  = blockM + ldRow;
    const int bRow  = blockN + ldRow;
    const bool aValid = (aRow < M);
    const _Float16* aSrc = A  + (size_t)aRow * K + ldCol;
    const _Float16* bSrc = Bt + (size_t)bRow * K + ldCol;
    const int ldsOff = ldRow * LDB + ldCol;

    v8f acc[4][2];
    for (int i = 0; i < 4; i++)
        for (int j = 0; j < 2; j++)
            for (int e = 0; e < 8; e++) acc[i][j][e] = 0.0f;

    v8h zz;
    for (int e = 0; e < 8; e++) zz[e] = (_Float16)0.f;

    const int nk = K / BK;
    v8h ra0, ra1, rb0, rb1;

    // prologue: stage tile 0 into buffer 0
    if (aValid) { const v8h* s = (const v8h*)aSrc; ra0 = s[0]; ra1 = s[1]; }
    else        { ra0 = zz; ra1 = zz; }
    { const v8h* s = (const v8h*)bSrc; rb0 = s[0]; rb1 = s[1]; }
    { v8h* d = (v8h*)&As[0][ldsOff]; d[0] = ra0; d[1] = ra1; }
    { v8h* d = (v8h*)&Bs[0][ldsOff]; d[0] = rb0; d[1] = rb1; }
    __syncthreads();

    for (int i = 0; i < nk; i++) {
        const int cur = i & 1;
        const bool haveNext = (i + 1) < nk;

        // issue next tile's global loads before the WMMA block
        if (haveNext) {
            const int k1 = (i + 1) * BK;
            if (aValid) { const v8h* s = (const v8h*)(aSrc + k1); ra0 = s[0]; ra1 = s[1]; }
            else        { ra0 = zz; ra1 = zz; }
            { const v8h* s = (const v8h*)(bSrc + k1); rb0 = s[0]; rb1 = s[1]; }
            if (i + 2 < nk) {                       // pull tile i+2 toward the caches
                if (aValid) __builtin_prefetch(aSrc + (i + 2) * BK, 0, 3);
                __builtin_prefetch(bSrc + (i + 2) * BK, 0, 3);
            }
        }

        // compute from current buffer
        v16h afr[4], bfr[2];
        for (int ni = 0; ni < 2; ni++) {
            const _Float16* p = &Bs[cur][(wnBase + ni * 16 + lr) * LDB];
            union { v16h v; v8h h[2]; } u;
            u.h[0] = *(const v8h*)(p + kB0);
            u.h[1] = *(const v8h*)(p + kB0 + 16);
            bfr[ni] = u.v;
        }
        for (int mi = 0; mi < 4; mi++) {
            const _Float16* p = &As[cur][(wmBase + mi * 16 + lr) * LDB];
            union { v16h v; v8h h[2]; } u;
            u.h[0] = *(const v8h*)(p + kA0);
            u.h[1] = *(const v8h*)(p + kA1);
            afr[mi] = u.v;
        }
        for (int mi = 0; mi < 4; mi++)
            for (int ni = 0; ni < 2; ni++)
                acc[mi][ni] = __builtin_amdgcn_wmma_f32_16x16x32_f16(
                    false, afr[mi], false, bfr[ni], (short)0, acc[mi][ni], false, false);

        // stage next tile into the other buffer (no conflict with readers of cur)
        if (haveNext) {
            const int nxt = cur ^ 1;
            v8h* dA = (v8h*)&As[nxt][ldsOff]; dA[0] = ra0; dA[1] = ra1;
            v8h* dB = (v8h*)&Bs[nxt][ldsOff]; dB[0] = rb0; dB[1] = rb1;
        }
        __syncthreads();
    }

    const float al = alphap[0];
    for (int mi = 0; mi < 4; mi++) {
        for (int ni = 0; ni < 2; ni++) {
            const int gcol = blockN + wnBase + ni * 16 + lr;
            for (int r = 0; r < 8; r++) {
                const int grow = blockM + wmBase + mi * 16 + hs * 8 + r;
                if (grow >= M) continue;
                const size_t o = (size_t)grow * Nn + gcol;
                const float g = acc[mi][ni][r] + bias[gcol];
                if (EPI == 0) {
                    outF[o] = g;
                } else if (EPI == 1) {
                    outF[o] = resid[o] + al * g;
                } else if (EPI == 2) {
                    outH[o] = (_Float16)(0.5f * g * (1.0f + erff(g * 0.70710678118f)));
                } else {
                    outF[o] = resid[o] + g;
                }
            }
        }
    }
}

// ---------------- attention: per pixel, 8 heads, top-5 softmax over T=77 ----
__global__ __launch_bounds__(256)
void attention_kernel(const float* __restrict__ q, const float* __restrict__ k,
                      const float* __restrict__ v, const int* __restrict__ mask,
                      const int* __restrict__ allpad,
                      const float* __restrict__ logit_scale,
                      _Float16* __restrict__ aligned16) {
    __shared__ float qs[CV];
    __shared__ float sims[HEADS * 80];
    __shared__ float att [HEADS * 80];
    __shared__ float thr[HEADS], mxv[HEADS], inv[HEADS];

    const int pix  = blockIdx.x;
    const int b    = pix >> 12;            // /4096
    const int tid  = threadIdx.x;
    const int lane = tid & 31;
    const int wave = tid >> 5;             // == head

    // 1) load q row + per-head l2norm
    {
        const int e0 = wave * DH + lane * 2;
        float q0 = q[(size_t)pix * CV + e0];
        float q1 = q[(size_t)pix * CV + e0 + 1];
        float ss = wave_sum(q0 * q0 + q1 * q1);
        float iv = 1.0f / fmaxf(sqrtf(ss), 1e-6f);
        qs[e0] = q0 * iv; qs[e0 + 1] = q1 * iv;
    }
    float ls = logit_scale[0];
    ls = fminf(fmaxf(ls, -2.0f), 2.0f);
    const float scale = __expf(ls) * 0.125f;   // * 1/sqrt(64)
    __syncthreads();

    // 2) sim[h][t]
    for (int j = tid; j < HEADS * TT; j += 256) {
        const int h = j / TT, t = j - h * TT;
        const float* kr = k + ((size_t)(b * TT + t)) * CV + h * DH;
        const float* qr = qs + h * DH;
        float d = 0.f;
        for (int c = 0; c < DH; c++) d += qr[c] * kr[c];
        float sv = d * scale;
        if (mask[b * TT + t]) sv = -1e30f;
        sims[h * 80 + t] = sv;
    }
    __syncthreads();

    // 3) top-5 threshold + max per head
    if (tid < HEADS) {
        float t1 = -1e30f, t2 = -1e30f, t3 = -1e30f, t4 = -1e30f, t5 = -1e30f;
        for (int t = 0; t < TT; t++) {
            float v0 = sims[tid * 80 + t];
            if      (v0 > t1) { t5 = t4; t4 = t3; t3 = t2; t2 = t1; t1 = v0; }
            else if (v0 > t2) { t5 = t4; t4 = t3; t3 = t2; t2 = v0; }
            else if (v0 > t3) { t5 = t4; t4 = t3; t3 = v0; }
            else if (v0 > t4) { t5 = t4; t4 = v0; }
            else if (v0 > t5) { t5 = v0; }
        }
        thr[tid] = t5; mxv[tid] = t1;
    }
    __syncthreads();

    // 4) masked exp
    for (int j = tid; j < HEADS * TT; j += 256) {
        const int h = j / TT, t = j - h * TT;
        const float sv = sims[h * 80 + t];
        att[h * 80 + t] = (sv >= thr[h] && sv > -1e29f) ? __expf(sv - mxv[h]) : 0.f;
    }
    __syncthreads();

    // 5) per-head sums
    if (tid < HEADS) {
        float s = 0.f;
        for (int t = 0; t < TT; t++) s += att[tid * 80 + t];
        inv[tid] = (allpad[b] || s <= 0.f) ? 0.f : 1.0f / s;
    }
    __syncthreads();

    // 6) aligned = attn @ v (f16 out for next GEMM)
    for (int j = tid; j < CV; j += 256) {
        const int h = j >> 6, d0 = j & 63;
        const float* vr = v + ((size_t)(b * TT)) * CV + h * DH + d0;
        float a = 0.f;
        for (int t = 0; t < TT; t++) a += att[h * 80 + t] * vr[(size_t)t * CV];
        aligned16[(size_t)pix * CV + j] = (_Float16)(a * inv[h]);
    }
}

// ---------------- host orchestration ----------------------------------------
extern "C" void kernel_launch(void* const* d_in, const int* in_sizes, int n_in,
                              void* d_out, int out_size, void* d_ws, size_t ws_size,
                              hipStream_t stream) {
    const float* visual   = (const float*)d_in[0];
    const float* text     = (const float*)d_in[1];
    const float* ln1_g    = (const float*)d_in[2];
    const float* ln1_b    = (const float*)d_in[3];
    const float* wq       = (const float*)d_in[4];
    const float* bq       = (const float*)d_in[5];
    const float* wk       = (const float*)d_in[6];
    const float* bk       = (const float*)d_in[7];
    const float* wv       = (const float*)d_in[8];
    const float* bv       = (const float*)d_in[9];
    const float* wo       = (const float*)d_in[10];
    const float* bo       = (const float*)d_in[11];
    const float* ln2_g    = (const float*)d_in[12];
    const float* ln2_b    = (const float*)d_in[13];
    const float* w1       = (const float*)d_in[14];
    const float* b1       = (const float*)d_in[15];
    const float* w2       = (const float*)d_in[16];
    const float* b2       = (const float*)d_in[17];
    const float* alpha    = (const float*)d_in[18];
    const float* logit_sc = (const float*)d_in[19];
    float* out = (float*)d_out;
    char*  ws  = (char*)d_ws;

    // workspace layout (bytes)
    float*    x_f32  = (float*)   (ws + 0);                    //  67,108,864
    _Float16* x16    = (_Float16*)(ws + 67108864ull);          //  33,554,432
    float*    q_f32  = (float*)   (ws + 100663296ull);         //  67MB (aliases h16)
    _Float16* h16    = (_Float16*)(ws + 100663296ull);         // 134,217,728
    _Float16* al16   = (_Float16*)(ws + 234881024ull);         //  33,554,432 (then y16)
    _Float16* y16    = al16;
    _Float16* wq16t  = (_Float16*)(ws + 268435456ull);
    _Float16* wk16t  = (_Float16*)(ws + 268959744ull);
    _Float16* wv16t  = (_Float16*)(ws + 269484032ull);
    _Float16* wo16t  = (_Float16*)(ws + 270008320ull);
    _Float16* w116t  = (_Float16*)(ws + 270532608ull);
    _Float16* w216t  = (_Float16*)(ws + 272629760ull);
    _Float16* text16 = (_Float16*)(ws + 274726912ull);
    float*    k_f32  = (float*)   (ws + 275357696ull);
    float*    v_f32  = (float*)   (ws + 276619264ull);
    int*      mask   = (int*)     (ws + 277880832ull);
    int*      allpad = (int*)     (ws + 277883392ull);

    // weight transposes + converts
    {
        int tot = CV * CV;
        transpose_cvt_kernel<<<(tot + 255) / 256, 256, 0, stream>>>(wq, wq16t, CV, CV);
        transpose_cvt_kernel<<<(tot + 255) / 256, 256, 0, stream>>>(wk, wk16t, CV, CV);
        transpose_cvt_kernel<<<(tot + 255) / 256, 256, 0, stream>>>(wv, wv16t, CV, CV);
        transpose_cvt_kernel<<<(tot + 255) / 256, 256, 0, stream>>>(wo, wo16t, CV, CV);
        int tot1 = CV * FF;
        transpose_cvt_kernel<<<(tot1 + 255) / 256, 256, 0, stream>>>(w1, w116t, CV, FF);
        transpose_cvt_kernel<<<(tot1 + 255) / 256, 256, 0, stream>>>(w2, w216t, FF, CV);
        int ct = KV_ROWS * CV;
        cvt_kernel<<<(ct + 255) / 256, 256, 0, stream>>>(text, text16, ct);
    }
    // pad masks
    padmask_kernel<<<KV_ROWS, 128, 0, stream>>>(text, mask);
    allpad_kernel<<<1, 32, 0, stream>>>(mask, allpad);

    // LN1 -> x (f32 + f16)
    layernorm_kernel<<<ROWS, 128, 0, stream>>>(visual, ln1_g, ln1_b, x_f32, x16);

    // q = x @ wq + bq
    gemm_wmma_f16<0><<<dim3(CV / 128, ROWS / 128), 256, 0, stream>>>(
        x16, wq16t, bq, x_f32, alpha, q_f32, h16, ROWS, CV, CV);
    // k, v = text @ wk/wv + b
    gemm_wmma_f16<0><<<dim3(CV / 128, (KV_ROWS + 127) / 128), 256, 0, stream>>>(
        text16, wk16t, bk, x_f32, alpha, k_f32, h16, KV_ROWS, CV, CV);
    gemm_wmma_f16<0><<<dim3(CV / 128, (KV_ROWS + 127) / 128), 256, 0, stream>>>(
        text16, wv16t, bv, x_f32, alpha, v_f32, h16, KV_ROWS, CV, CV);

    // l2norm k per (b,t,h)
    l2norm_heads_kernel<<<(KV_ROWS * HEADS + 7) / 8, 256, 0, stream>>>(k_f32, KV_ROWS * HEADS);

    // attention -> aligned16
    attention_kernel<<<ROWS, 256, 0, stream>>>(q_f32, k_f32, v_f32, mask, allpad,
                                               logit_sc, al16);

    // ypre = x + alpha*(aligned @ wo + bo)   (written into d_out)
    gemm_wmma_f16<1><<<dim3(CV / 128, ROWS / 128), 256, 0, stream>>>(
        al16, wo16t, bo, x_f32, alpha, out, h16, ROWS, CV, CV);

    // LN2 in place on d_out, also emit y16
    layernorm_kernel<<<ROWS, 128, 0, stream>>>(out, ln2_g, ln2_b, out, y16);

    // h = gelu(y @ w1 + b1)  (f16)
    gemm_wmma_f16<2><<<dim3(FF / 128, ROWS / 128), 256, 0, stream>>>(
        y16, w116t, b1, out, alpha, out, h16, ROWS, FF, CV);

    // out = y + h @ w2 + b2
    gemm_wmma_f16<3><<<dim3(CV / 128, ROWS / 128), 256, 0, stream>>>(
        h16, w216t, b2, out, alpha, out, h16, ROWS, CV, FF);
}
